// RvNN_8701603741792
// MI455X (gfx1250) — matrix operand order repaired and checked by reading, not compiled, not run on hardware
//
#include <hip/hip_runtime.h>
#include <hip/hip_bf16.h>
#include <math.h>

#define N_NODES 4096
#define WORDS   64
#define HID     256
#define VOCAB   50257
#define NCLASS  4
#define N_LEAF  2048
#define SCAN_B  32   // sub-batch rows per level pass (2 row-tiles x 16 col-tiles = 32 wmma tiles = 32 waves)

typedef __attribute__((ext_vector_type(16))) _Float16 v16h;
typedef __attribute__((ext_vector_type(8)))  _Float16 v8h;
typedef __attribute__((ext_vector_type(8)))  float    v8f;

union V16 { v16h v; v8h h[2]; };

// ---- WMMA fragment loaders (layouts per CDNA5 ISA 7.12.2, wave32) ----
// A: 16x32 f16 tile of row-major matrix (lda halves). lane<16: rows M=lane, K={0..7,16..23};
// lane>=16: rows M=lane-16, K={8..15,24..31}.
__device__ __forceinline__ v16h load_a_frag(const _Float16* __restrict__ A, int lda,
                                            int m0, int k0, int lane) {
  int row  = m0 + (lane & 15);
  int koff = k0 + ((lane >= 16) ? 8 : 0);
  const _Float16* p = A + (size_t)row * lda + koff;
  V16 u;
  u.h[0] = *(const v8h*)(p);        // K = koff .. koff+7
  u.h[1] = *(const v8h*)(p + 16);   // K = koff+16 .. koff+23
  return u.v;
}

// B: 32x16 f16 tile where B[k][n] = Wt[n][k] (Wt row-major N x K).
// lane<16: N = lane, K = k0+{0..7,16..23}; lane>=16: N = lane-16, K = k0+{8..15,24..31}.
// => two 16-byte runs per lane from row n of Wt (matches A fragment order).
__device__ __forceinline__ v16h load_b_frag(const _Float16* __restrict__ Wt, int ldk,
                                            int n0, int k0, int lane) {
  int n    = n0 + (lane & 15);
  int koff = k0 + ((lane >= 16) ? 8 : 0);
  const _Float16* p = Wt + (size_t)n * ldk + koff;
  V16 u;
  u.h[0] = *(const v8h*)(p);
  u.h[1] = *(const v8h*)(p + 16);
  return u.v;
}

__device__ __forceinline__ v8f wmma_f16(v16h a, v16h b, v8f c) {
  return __builtin_amdgcn_wmma_f32_16x16x32_f16(false, a, false, b, (short)0, c, false, false);
}

__device__ __forceinline__ float sigmoidf_(float x) { return 1.0f / (1.0f + expf(-x)); }

// ---------------- Kernel 0: tiled transpose  Et[v][h] = E[h][v] ----------------
// Pays the strided access pattern exactly once; all embedding gathers become coalesced rows.
__global__ __launch_bounds__(256) void transpose_E(const float* __restrict__ E,
                                                   float* __restrict__ Et) {
  __shared__ float tile[32][33];
  int v0 = blockIdx.x * 32;
  int h0 = blockIdx.y * 32;
  int tx = threadIdx.x & 31, ty = threadIdx.x >> 5;   // 32 x 8
  #pragma unroll
  for (int j = ty; j < 32; j += 8) {
    int v = v0 + tx;
    tile[j][tx] = (v < VOCAB) ? E[(size_t)(h0 + j) * VOCAB + v] : 0.0f;
  }
  __syncthreads();
  #pragma unroll
  for (int j = ty; j < 32; j += 8) {
    int v = v0 + j;
    if (v < VOCAB) Et[(size_t)v * HID + (h0 + tx)] = tile[tx][j];
  }
}

// ---------------- Kernel 1a: gather-sum from transposed table (coalesced) ----------------
__global__ __launch_bounds__(HID) void embed_gather_t(const int* __restrict__ tree,
                                                      const float* __restrict__ Et,
                                                      float* __restrict__ X) {
  __shared__ int toks[WORDS];
  int node = blockIdx.x;
  if (threadIdx.x < WORDS) toks[threadIdx.x] = tree[node * WORDS + threadIdx.x];
  __syncthreads();
  int h = threadIdx.x;
  float acc = 0.0f;
  #pragma unroll 8
  for (int w = 0; w < WORDS; ++w) acc += Et[(size_t)toks[w] * HID + h];
  X[(size_t)node * HID + h] = acc;
}

// ---------------- Kernel 1b: fallback direct gather (if workspace too small for Et) ----------------
__global__ __launch_bounds__(HID) void embed_gather(const int* __restrict__ tree,
                                                    const float* __restrict__ E,
                                                    float* __restrict__ X) {
  __shared__ int toks[WORDS];
  int node = blockIdx.x;
  if (threadIdx.x < WORDS) toks[threadIdx.x] = tree[node * WORDS + threadIdx.x];
  __syncthreads();
  int h = threadIdx.x;
  const float* Erow = E + (size_t)h * VOCAB;
  float acc = 0.0f;
  #pragma unroll 8
  for (int w = 0; w < WORDS; ++w) acc += Erow[toks[w]];
  X[(size_t)node * HID + h] = acc;
}

// ---------------- Kernel 2: f32 -> f16 convert ----------------
__global__ void f32_to_f16(const float* __restrict__ src, _Float16* __restrict__ dst, int n) {
  int i = blockIdx.x * blockDim.x + threadIdx.x;
  if (i < n) dst[i] = (_Float16)src[i];
}

// ---------------- Kernel 3: fused triple WMMA GEMM  O{z,r,h} = X @ W{z,r,h}^T ----------------
// grid: (M/16, N/(16*8)), block 256 (8 waves, one 16x16 C tile each).
// Each A fragment is shared by three v_wmma issues (z, r, h accumulators).
__global__ __launch_bounds__(256) void wmma_gemm3(
    const _Float16* __restrict__ Xh,
    const _Float16* __restrict__ Wzh, const _Float16* __restrict__ Wrh, const _Float16* __restrict__ Whh,
    float* __restrict__ Oz, float* __restrict__ Or, float* __restrict__ Oh) {
  int lane = threadIdx.x & 31, wave = threadIdx.x >> 5;
  int m0 = blockIdx.x * 16;
  int n0 = (blockIdx.y * 8 + wave) * 16;
  v8f accz = {}, accr = {}, acch = {};
  #pragma unroll
  for (int k0 = 0; k0 < HID; k0 += 32) {
    v16h a  = load_a_frag(Xh, HID, m0, k0, lane);
    v16h bz = load_b_frag(Wzh, HID, n0, k0, lane);
    accz = wmma_f16(a, bz, accz);
    v16h br = load_b_frag(Wrh, HID, n0, k0, lane);
    accr = wmma_f16(a, br, accr);
    v16h bh = load_b_frag(Whh, HID, n0, k0, lane);
    acch = wmma_f16(a, bh, acch);
  }
  int col   = n0 + (lane & 15);
  int rbase = m0 + ((lane >= 16) ? 8 : 0);
  #pragma unroll
  for (int v = 0; v < 8; ++v) {
    size_t idx = (size_t)(rbase + v) * HID + col;
    Oz[idx] = accz[v];
    Or[idx] = accr[v];
    Oh[idx] = acch[v];
  }
}

// ---------------- Kernel 4: tree depths, level offsets, level-sorted order (1 thread) ----------------
__global__ void build_levels(const int* __restrict__ edge, int* __restrict__ depth,
                             int* __restrict__ cnt, int* __restrict__ off,
                             int* __restrict__ order, int* __restrict__ nlev) {
  if (threadIdx.x != 0 || blockIdx.x != 0) return;
  for (int i = 0; i < N_NODES; ++i) cnt[i] = 0;
  depth[0] = 0; cnt[0] = 1;
  int maxd = 0;
  for (int i = 1; i < N_NODES; ++i) {
    int d = depth[edge[2 * i]] + 1;
    depth[i] = d;
    cnt[d]++;
    if (d > maxd) maxd = d;
  }
  off[0] = 0;
  for (int l = 0; l <= maxd; ++l) off[l + 1] = off[l] + cnt[l];
  for (int l = 0; l <= maxd; ++l) cnt[l] = 0;
  for (int i = 0; i < N_NODES; ++i) {           // stable: ascending node id within level
    int d = depth[i];
    order[off[d] + cnt[d]] = i;
    cnt[d]++;
  }
  *nlev = maxd + 1;
}

// ---------------- Kernel 5: h[0] = X[0] ----------------
__global__ void init_h0(const float* __restrict__ X, float* __restrict__ node_h) {
  node_h[threadIdx.x] = X[threadIdx.x];
}

// ---------------- Kernel 6: persistent level-wavefront GRU scan (1 workgroup, 32 waves) ----------------
__global__ __launch_bounds__(1024) void tree_gru_scan(
    const int* __restrict__ edge, const int* __restrict__ order,
    const int* __restrict__ off, const int* __restrict__ nlev_p,
    const _Float16* __restrict__ Uzh, const _Float16* __restrict__ Urh, const _Float16* __restrict__ Uhh,
    const float* __restrict__ WzX, const float* __restrict__ WrX, const float* __restrict__ WhX,
    const float* __restrict__ b_z, const float* __restrict__ b_r, const float* __restrict__ b_h,
    float* __restrict__ node_h) {
  __shared__ alignas(16) _Float16 s_ph [SCAN_B * HID];  // parent h (f16, WMMA A)
  __shared__ alignas(16) _Float16 s_phr[SCAN_B * HID];  // r * parent h (f16, WMMA A)
  __shared__ alignas(16) _Float16 s_z  [SCAN_B * HID];  // update gate
  __shared__ int s_nid[SCAN_B];
  __shared__ int s_par[SCAN_B];

  int tid  = threadIdx.x;
  int lane = tid & 31, wave = tid >> 5;
  int nlev = nlev_p[0];

  // Pull the recurrent weights (3 x 128 KB f16) toward this WGP's caches once;
  // they are re-read every level of the serial chain. Emits global_prefetch_b8.
  {
    const size_t bytes = (size_t)HID * HID * sizeof(_Float16);
    for (size_t o = (size_t)tid * 128; o < bytes; o += 1024ull * 128) {
      __builtin_prefetch((const char*)Uzh + o, 0, 0);
      __builtin_prefetch((const char*)Urh + o, 0, 0);
      __builtin_prefetch((const char*)Uhh + o, 0, 0);
    }
  }

  for (int lvl = 1; lvl < nlev; ++lvl) {
    int base = off[lvl];
    int cntl = off[lvl + 1] - base;
    for (int sb = 0; sb < cntl; sb += SCAN_B) {
      int Bn = cntl - sb; if (Bn > SCAN_B) Bn = SCAN_B;

      // ---- stage parent states into LDS ----
      if (tid < SCAN_B) {
        int nid = (tid < Bn) ? order[base + sb + tid] : -1;
        s_nid[tid] = nid;
        s_par[tid] = (nid >= 0) ? edge[2 * nid] : 0;
      }
      __syncthreads();
      for (int e = tid; e < SCAN_B * HID; e += 1024) {
        int m = e >> 8;
        float v = (m < Bn) ? node_h[(size_t)s_par[m] * HID + (e & 255)] : 0.0f;
        s_ph[e]  = (_Float16)v;
        s_phr[e] = (_Float16)0.0f;   // pre-zero (pads stay zero for WMMA)
      }
      __syncthreads();

      // ---- pass 1: z = sig(WzX + Uz@ph + bz), r = sig(WrX + Ur@ph + br), phr = r*ph ----
      {
        int mt = wave >> 4, nt = wave & 15;      // 2 x 16 tiles == 32 waves
        int m0 = mt * 16, n0 = nt * 16;
        v8f accz = {}, accr = {};
        #pragma unroll
        for (int k0 = 0; k0 < HID; k0 += 32) {
          v16h a  = load_a_frag(s_ph, HID, m0, k0, lane);
          v16h bz = load_b_frag(Uzh, HID, n0, k0, lane);
          accz = wmma_f16(a, bz, accz);
          v16h br = load_b_frag(Urh, HID, n0, k0, lane);
          accr = wmma_f16(a, br, accr);
        }
        int col   = n0 + (lane & 15);
        int rbase = m0 + ((lane >= 16) ? 8 : 0);
        #pragma unroll
        for (int v = 0; v < 8; ++v) {
          int m = rbase + v;
          if (m < Bn) {
            int nid = s_nid[m];
            float z = sigmoidf_(accz[v] + WzX[(size_t)nid * HID + col] + b_z[col]);
            float r = sigmoidf_(accr[v] + WrX[(size_t)nid * HID + col] + b_r[col]);
            s_z[m * HID + col]   = (_Float16)z;
            s_phr[m * HID + col] = (_Float16)(r * (float)s_ph[m * HID + col]);
          }
        }
      }
      __syncthreads();

      // ---- pass 2: c = tanh(WhX + Uh@(r*ph) + bh); h = z*ph + (1-z)*c ----
      {
        int mt = wave >> 4, nt = wave & 15;
        int m0 = mt * 16, n0 = nt * 16;
        v8f acch = {};
        #pragma unroll
        for (int k0 = 0; k0 < HID; k0 += 32) {
          v16h a  = load_a_frag(s_phr, HID, m0, k0, lane);
          v16h bh = load_b_frag(Uhh, HID, n0, k0, lane);
          acch = wmma_f16(a, bh, acch);
        }
        int col   = n0 + (lane & 15);
        int rbase = m0 + ((lane >= 16) ? 8 : 0);
        #pragma unroll
        for (int v = 0; v < 8; ++v) {
          int m = rbase + v;
          if (m < Bn) {
            int nid = s_nid[m];
            float c  = tanhf(acch[v] + WhX[(size_t)nid * HID + col] + b_h[col]);
            float z  = (float)s_z[m * HID + col];
            float ph = node_h[(size_t)s_par[m] * HID + col];   // full-precision parent
            node_h[(size_t)nid * HID + col] = z * ph + (1.0f - z) * c;
          }
        }
      }
      __threadfence();   // make h visible to next level's LDS staging
      __syncthreads();
    }
  }
}

// ---------------- Kernel 7: leaf max-pool + softmax + loss ----------------
__global__ __launch_bounds__(HID) void leaf_softmax(
    const int* __restrict__ leaf_idxs, const float* __restrict__ node_h,
    const float* __restrict__ W_out, const float* __restrict__ b_out,
    const float* __restrict__ y, float* __restrict__ out) {
  __shared__ float fs[HID];
  __shared__ float logits[NCLASS];
  int h = threadIdx.x;
  float m = -1e30f;
  for (int i = 0; i < N_LEAF; ++i)
    m = fmaxf(m, node_h[(size_t)leaf_idxs[i] * HID + h]);
  fs[h] = m;
  __syncthreads();
  if (h < NCLASS) {
    float acc = b_out[h];
    for (int k = 0; k < HID; ++k) acc += W_out[h * HID + k] * fs[k];
    logits[h] = acc;
  }
  __syncthreads();
  if (h == 0) {
    float mx = logits[0];
    for (int i = 1; i < NCLASS; ++i) mx = fmaxf(mx, logits[i]);
    float e[NCLASS], s = 0.0f;
    for (int i = 0; i < NCLASS; ++i) { e[i] = expf(logits[i] - mx); s += e[i]; }
    float loss = 0.0f;
    for (int i = 0; i < NCLASS; ++i) {
      float p = e[i] / s;
      out[i] = p;
      float d = y[i] - p;
      loss += d * d;
    }
    out[NCLASS] = loss;
  }
}

extern "C" void kernel_launch(void* const* d_in, const int* in_sizes, int n_in,
                              void* d_out, int out_size, void* d_ws, size_t ws_size,
                              hipStream_t stream) {
  const int*   tree      = (const int*)  d_in[0];
  const int*   edge      = (const int*)  d_in[1];
  const int*   leaf_idxs = (const int*)  d_in[2];
  const float* y         = (const float*)d_in[3];
  const float* E         = (const float*)d_in[4];
  const float* W_z       = (const float*)d_in[5];
  const float* U_z       = (const float*)d_in[6];
  const float* b_z       = (const float*)d_in[7];
  const float* W_r       = (const float*)d_in[8];
  const float* U_r       = (const float*)d_in[9];
  const float* b_r       = (const float*)d_in[10];
  const float* W_h       = (const float*)d_in[11];
  const float* U_h       = (const float*)d_in[12];
  const float* b_h       = (const float*)d_in[13];
  const float* W_out     = (const float*)d_in[14];
  const float* b_out     = (const float*)d_in[15];
  float* out = (float*)d_out;

  // ---- workspace carve-up (256B aligned) ----
  char* w = (char*)d_ws;
  auto alloc = [&](size_t bytes) -> char* {
    char* p = w;
    w += (bytes + 255) & ~(size_t)255;
    return p;
  };
  float*    X      = (float*)   alloc((size_t)N_NODES * HID * 4);
  float*    WzX    = (float*)   alloc((size_t)N_NODES * HID * 4);
  float*    WrX    = (float*)   alloc((size_t)N_NODES * HID * 4);
  float*    WhX    = (float*)   alloc((size_t)N_NODES * HID * 4);
  float*    node_h = (float*)   alloc((size_t)N_NODES * HID * 4);
  _Float16* Xh     = (_Float16*)alloc((size_t)N_NODES * HID * 2);
  _Float16* Wzh    = (_Float16*)alloc((size_t)HID * HID * 2);
  _Float16* Wrh    = (_Float16*)alloc((size_t)HID * HID * 2);
  _Float16* Whh    = (_Float16*)alloc((size_t)HID * HID * 2);
  _Float16* Uzh    = (_Float16*)alloc((size_t)HID * HID * 2);
  _Float16* Urh    = (_Float16*)alloc((size_t)HID * HID * 2);
  _Float16* Uhh    = (_Float16*)alloc((size_t)HID * HID * 2);
  int*      depth  = (int*)     alloc((size_t)N_NODES * 4);
  int*      cnt    = (int*)     alloc((size_t)N_NODES * 4);
  int*      off    = (int*)     alloc((size_t)(N_NODES + 1) * 4);
  int*      order  = (int*)     alloc((size_t)N_NODES * 4);
  int*      nlev   = (int*)     alloc(256);
  (void)n_in; (void)in_sizes; (void)out_size;

  // Optional transposed embedding table (51.5 MB) if workspace allows:
  // turns 67M scattered 4B gathers into coalesced 1KB row reads.
  size_t used    = (size_t)(w - (char*)d_ws);
  size_t etBytes = (size_t)VOCAB * HID * sizeof(float);
  bool   useEt   = (ws_size >= used + etBytes + 256);

  // 1) X = gather-sum of embeddings
  if (useEt) {
    float* Et = (float*)alloc(etBytes);
    transpose_E<<<dim3((VOCAB + 31) / 32, HID / 32), 256, 0, stream>>>(E, Et);
    embed_gather_t<<<N_NODES, HID, 0, stream>>>(tree, Et, X);
  } else {
    embed_gather<<<N_NODES, HID, 0, stream>>>(tree, E, X);
  }

  // 2) f16 copies for WMMA operands
  f32_to_f16<<<(N_NODES * HID + 255) / 256, 256, 0, stream>>>(X,   Xh,  N_NODES * HID);
  f32_to_f16<<<(HID * HID + 255) / 256,   256, 0, stream>>>(W_z, Wzh, HID * HID);
  f32_to_f16<<<(HID * HID + 255) / 256,   256, 0, stream>>>(W_r, Wrh, HID * HID);
  f32_to_f16<<<(HID * HID + 255) / 256,   256, 0, stream>>>(W_h, Whh, HID * HID);
  f32_to_f16<<<(HID * HID + 255) / 256,   256, 0, stream>>>(U_z, Uzh, HID * HID);
  f32_to_f16<<<(HID * HID + 255) / 256,   256, 0, stream>>>(U_r, Urh, HID * HID);
  f32_to_f16<<<(HID * HID + 255) / 256,   256, 0, stream>>>(U_h, Uhh, HID * HID);

  // 3) WzX / WrX / WhX = X @ W^T via fused WMMA (A fragments shared across z/r/h)
  wmma_gemm3<<<dim3(N_NODES / 16, HID / (16 * 8)), 256, 0, stream>>>(
      Xh, Wzh, Wrh, Whh, WzX, WrX, WhX);

  // 4) level decomposition of the tree (wavefront parallelism)
  build_levels<<<1, 1, 0, stream>>>(edge, depth, cnt, off, order, nlev);

  // 5) h[0] = X[0]
  init_h0<<<1, HID, 0, stream>>>(X, node_h);

  // 6) level-by-level batched GRU on one persistent workgroup (WMMA + LDS)
  tree_gru_scan<<<1, 1024, 0, stream>>>(edge, order, off, nlev,
                                        Uzh, Urh, Uhh, WzX, WrX, WhX,
                                        b_z, b_r, b_h, node_h);

  // 7) leaf max-pool, softmax, loss
  leaf_softmax<<<1, HID, 0, stream>>>(leaf_idxs, node_h, W_out, b_out, y, out);
}
